// AttentionBlock_35390530519683
// MI455X (gfx1250) — matrix operand order repaired and checked by reading, not compiled, not run on hardware
//
#include <hip/hip_runtime.h>
#include <hip/hip_bf16.h>

typedef _Float16 f16;
typedef _Float16 v16h __attribute__((ext_vector_type(16)));
typedef _Float16 v8h  __attribute__((ext_vector_type(8)));
typedef float    v8f  __attribute__((ext_vector_type(8)));

#define NB     4
#define NC     128
#define NHW    4096
#define NHEADS 4
#define NHD    32
#define NGROUP 32

// ---------------- WMMA helpers (CDNA5, wave32) ----------------

static __device__ __forceinline__ v8f wmma32(v16h a, v16h b, v8f c) {
  // v_wmma_f32_16x16x32_f16: D = A(16x32) * B(32x16) + C(16x16)
  return __builtin_amdgcn_wmma_f32_16x16x32_f16(false, a, false, b, (short)0, c,
                                                false, false);
}

// A fragment (16x32 f16). rowp = base + (lane&15)*ld, rows contiguous in K.
// Lanes 0-15 hold K 0..15, lanes 16-31 hold K 16..31 (matches load_bt32).
static __device__ __forceinline__ v16h load_a32(const f16* rowp, int lane, int kbase) {
  const int hi = lane >> 4;
  union { v16h v; v8h h[2]; } u;
  u.h[0] = *(const v8h*)(rowp + kbase + (hi ? 16 : 0));
  u.h[1] = *(const v8h*)(rowp + kbase + (hi ? 24 : 8));
  return u.v;
}

// B fragment (32x16 f16) sourced from Bt (N x K, row-major).
// btrow = Bt + (lane&15)*K. Lanes 0-15: K kbase..kbase+15; lanes 16-31: +16.
static __device__ __forceinline__ v16h load_bt32(const f16* btrow, int lane, int kbase) {
  return *(const v16h*)(btrow + kbase + ((lane >> 4) ? 16 : 0));
}

// reductions across the 16-lane half that holds one C/D row
static __device__ __forceinline__ float hmax16(float v) {
  v = fmaxf(v, __shfl_xor(v, 1, 32));
  v = fmaxf(v, __shfl_xor(v, 2, 32));
  v = fmaxf(v, __shfl_xor(v, 4, 32));
  v = fmaxf(v, __shfl_xor(v, 8, 32));
  return v;
}
static __device__ __forceinline__ float hsum16(float v) {
  v += __shfl_xor(v, 1, 32);
  v += __shfl_xor(v, 2, 32);
  v += __shfl_xor(v, 4, 32);
  v += __shfl_xor(v, 8, 32);
  return v;
}

// ---------------- preprocessing kernels ----------------

__global__ void cvt_f16_kernel(const float* __restrict__ src, f16* __restrict__ dst, int n) {
  int i = blockIdx.x * blockDim.x + threadIdx.x;
  if (i < n) dst[i] = (f16)src[i];
}

// one block per (b, group); group's 4 channels are contiguous: 16384 floats
__global__ __launch_bounds__(256) void gn_stats_kernel(const float* __restrict__ x,
                                                       float* __restrict__ stats) {
  __shared__ float ssum[256], ssq[256];
  const int bg = blockIdx.x;
  const float* p = x + (size_t)bg * 16384;
  float s = 0.f, q = 0.f;
  for (int i = threadIdx.x; i < 16384; i += 256) {
    float v = p[i];
    s += v; q += v * v;
  }
  ssum[threadIdx.x] = s; ssq[threadIdx.x] = q;
  __syncthreads();
  for (int st = 128; st > 0; st >>= 1) {
    if ((int)threadIdx.x < st) {
      ssum[threadIdx.x] += ssum[threadIdx.x + st];
      ssq[threadIdx.x]  += ssq[threadIdx.x + st];
    }
    __syncthreads();
  }
  if (threadIdx.x == 0) {
    float mean = ssum[0] * (1.0f / 16384.0f);
    float var  = ssq[0] * (1.0f / 16384.0f) - mean * mean;
    stats[2 * bg]     = mean;
    stats[2 * bg + 1] = rsqrtf(var + 1e-5f);
  }
}

// normalize + affine, transpose (B,C,HW) -> (B*HW, C) f16
__global__ __launch_bounds__(256) void gn_apply_kernel(const float* __restrict__ x,
                                                       const float* __restrict__ stats,
                                                       const float* __restrict__ w,
                                                       const float* __restrict__ b,
                                                       f16* __restrict__ xnt) {
  size_t idx = (size_t)blockIdx.x * 256 + threadIdx.x;  // < 4*128*4096
  int bi  = (int)(idx >> 19);
  int rem = (int)(idx & 524287);
  int c   = rem >> 12;
  int hw  = rem & 4095;
  int g   = c >> 2;
  float mean = stats[2 * (bi * NGROUP + g)];
  float rstd = stats[2 * (bi * NGROUP + g) + 1];
  float v = (x[idx] - mean) * rstd * w[c] + b[c];
  xnt[((size_t)bi * NHW + hw) * NC + c] = (f16)v;
}

// ---------------- QKV GEMM: (16384 x 128) @ (128 x 384) ----------------
// 16(M) x 64(N) per wave: A fragments loaded once, 16 WMMAs, 4 accumulators.
// Outputs routed to Q/K (bh,seq,hd) and Vt (bh,hd,seq).

__global__ __launch_bounds__(128) void qkv_gemm_kernel(const f16* __restrict__ xnt,
                                                       const f16* __restrict__ w16,
                                                       const float* __restrict__ bias,
                                                       f16* __restrict__ qbuf,
                                                       f16* __restrict__ kbuf,
                                                       f16* __restrict__ vtbuf) {
  const int lane = threadIdx.x & 31;
  const int wid  = blockIdx.x * 4 + (threadIdx.x >> 5);
  const int NG = (3 * NC) / 64;  // 6 groups of four 16-col tiles
  const int tg = wid % NG, tm = wid / NG;
  const int h = lane >> 4, nl = lane & 15;

  const f16* arow = xnt + (size_t)(tm * 16 + nl) * NC;
  const v16h a0 = load_a32(arow, lane, 0);
  const v16h a1 = load_a32(arow, lane, 32);
  const v16h a2 = load_a32(arow, lane, 64);
  const v16h a3 = load_a32(arow, lane, 96);

  v8f acc[4];
#pragma unroll
  for (int t = 0; t < 4; t++) {
    const f16* brow = w16 + (size_t)((tg * 4 + t) * 16 + nl) * NC;
    v8f c = {};
    c = wmma32(a0, load_bt32(brow, lane, 0), c);
    c = wmma32(a1, load_bt32(brow, lane, 32), c);
    c = wmma32(a2, load_bt32(brow, lane, 64), c);
    c = wmma32(a3, load_bt32(brow, lane, 96), c);
    acc[t] = c;
  }

#pragma unroll
  for (int t = 0; t < 4; t++) {
    const int ncol = (tg * 4 + t) * 16 + nl;
    const float bv = bias[ncol];
    const int region = ncol >> 7;      // 0=Q 1=K 2=V (uniform per tile)
    const int o    = ncol & (NC - 1);
    const int head = o >> 5, hd = o & (NHD - 1);
#pragma unroll
    for (int r = 0; r < 8; r++) {
      int row = tm * 16 + r + 8 * h;
      int b   = row >> 12, seq = row & (NHW - 1);
      size_t bh = (size_t)(b * NHEADS + head);
      f16 hv = (f16)(acc[t][r] + bv);
      if (region == 0)      qbuf[(bh * NHW + seq) * NHD + hd] = hv;
      else if (region == 1) kbuf[(bh * NHW + seq) * NHD + hd] = hv;
      else                  vtbuf[(bh * NHD + hd) * NHW + seq] = hv;
    }
  }
}

// ---------------- flash attention: one wave per 16-row q stripe ----------------
// 64 keys per online-softmax step; base-2 softmax with scale folded into Q.
// K fragments double-buffered and V fragments hoisted so global loads are in
// flight during the softmax VALU block.

__global__ __launch_bounds__(128) void attn_kernel(const f16* __restrict__ qbuf,
                                                   const f16* __restrict__ kbuf,
                                                   const f16* __restrict__ vtbuf,
                                                   f16* __restrict__ ot) {
  __shared__ __align__(16) f16 lds[4][16 * 72];  // per-wave 16x64 P tile, pitch 72
  const int lane = threadIdx.x & 31;
  const int wv   = threadIdx.x >> 5;
  const int wid  = blockIdx.x * 4 + wv;
  const int qt   = wid & 255;       // 4096/16 q tiles
  const int bh   = wid >> 8;        // 0..15
  const int nl = lane & 15, h = lane >> 4;

  const f16* Q  = qbuf  + (size_t)bh * NHW * NHD;
  const f16* K  = kbuf  + (size_t)bh * NHW * NHD;
  const f16* Vt = vtbuf + (size_t)bh * NHD * NHW;
  const f16* vrow0 = Vt + (size_t)nl * NHW;
  const f16* vrow1 = Vt + (size_t)(16 + nl) * NHW;
  f16* pt = &lds[wv][0];

  // scale * log2(e) folded into Q so softmax is pure exp2
  v16h qa = load_a32(Q + (size_t)(qt * 16 + nl) * NHD, lane, 0);
  const f16 qs = (f16)(0.17677669529663687f * 1.4426950408889634f);
#pragma unroll
  for (int j = 0; j < 16; j++) qa[j] *= qs;

  float mi[8], li[8];
#pragma unroll
  for (int r = 0; r < 8; r++) { mi[r] = -__builtin_inff(); li[r] = 0.f; }
  v8f o0 = {}, o1 = {};
  const v8f zc = {};

  // prologue: K fragments for chunk 0
  v16h kb0 = load_bt32(K + (size_t)(nl) * NHD, lane, 0);
  v16h kb1 = load_bt32(K + (size_t)(16 + nl) * NHD, lane, 0);
  v16h kb2 = load_bt32(K + (size_t)(32 + nl) * NHD, lane, 0);
  v16h kb3 = load_bt32(K + (size_t)(48 + nl) * NHD, lane, 0);

  for (int kt = 0; kt < NHW; kt += 64) {
    v8f s0 = wmma32(qa, kb0, zc);
    v8f s1 = wmma32(qa, kb1, zc);
    v8f s2 = wmma32(qa, kb2, zc);
    v8f s3 = wmma32(qa, kb3, zc);
    // V fragments for this chunk: independent of softmax, issue early
    v16h vb00 = load_bt32(vrow0, lane, kt);
    v16h vb01 = load_bt32(vrow0, lane, kt + 32);
    v16h vb10 = load_bt32(vrow1, lane, kt);
    v16h vb11 = load_bt32(vrow1, lane, kt + 32);
    // K fragments for the next chunk: hide under the softmax VALU block
    if (kt + 64 < NHW) {
      kb0 = load_bt32(K + (size_t)(kt + 64 + nl) * NHD, lane, 0);
      kb1 = load_bt32(K + (size_t)(kt + 80 + nl) * NHD, lane, 0);
      kb2 = load_bt32(K + (size_t)(kt + 96 + nl) * NHD, lane, 0);
      kb3 = load_bt32(K + (size_t)(kt + 112 + nl) * NHD, lane, 0);
    }
    if (kt + 128 < NHW) {
      // pull the chunk after that toward the WGP (global_prefetch_b8)
      __builtin_prefetch(K + (size_t)(kt + 128 + nl + 16 * h) * NHD, 0, 3);
      __builtin_prefetch(Vt + (size_t)(nl + 16 * h) * NHW + kt + 128, 0, 3);
    }
#pragma unroll
    for (int r = 0; r < 8; r++) {
      float mx = fmaxf(fmaxf(s0[r], s1[r]), fmaxf(s2[r], s3[r]));
      mx = hmax16(mx);
      float mnew = fmaxf(mi[r], mx);
      float corr = exp2f(mi[r] - mnew);
      float p0 = exp2f(s0[r] - mnew);
      float p1 = exp2f(s1[r] - mnew);
      float p2 = exp2f(s2[r] - mnew);
      float p3 = exp2f(s3[r] - mnew);
      li[r] = li[r] * corr + hsum16((p0 + p1) + (p2 + p3));
      mi[r] = mnew;
      o0[r] *= corr;
      o1[r] *= corr;
      f16* prow = pt + (r + 8 * h) * 72 + nl;
      prow[0]  = (f16)p0;
      prow[16] = (f16)p1;
      prow[32] = (f16)p2;
      prow[48] = (f16)p3;
    }
    // repack P (C/D layout) -> two A fragments via wave-private LDS tile
    v16h pa0 = load_a32(pt + nl * 72, lane, 0);
    v16h pa1 = load_a32(pt + nl * 72, lane, 32);
    o0 = wmma32(pa0, vb00, o0);
    o0 = wmma32(pa1, vb01, o0);
    o1 = wmma32(pa0, vb10, o1);
    o1 = wmma32(pa1, vb11, o1);
  }

  const int b = bh >> 2, head = bh & 3;
#pragma unroll
  for (int r = 0; r < 8; r++) {
    float inv = 1.0f / li[r];
    int seq = qt * 16 + r + 8 * h;
    size_t base = ((size_t)b * NHW + seq) * NC + head * NHD;
    ot[base + nl]      = (f16)(o0[r] * inv);
    ot[base + 16 + nl] = (f16)(o1[r] * inv);
  }
}

// ---------------- proj GEMM + bias + residual ----------------
// 16(M) x 64(N) per wave, same structure as qkv_gemm.

__global__ __launch_bounds__(128) void proj_gemm_kernel(const f16* __restrict__ ot,
                                                        const f16* __restrict__ w16,
                                                        const float* __restrict__ bias,
                                                        const float* __restrict__ skip,
                                                        float* __restrict__ out) {
  const int lane = threadIdx.x & 31;
  const int wid  = blockIdx.x * 4 + (threadIdx.x >> 5);
  const int NG = NC / 64;  // 2
  const int tg = wid % NG, tm = wid / NG;
  const int h = lane >> 4, nl = lane & 15;

  const f16* arow = ot + (size_t)(tm * 16 + nl) * NC;
  const v16h a0 = load_a32(arow, lane, 0);
  const v16h a1 = load_a32(arow, lane, 32);
  const v16h a2 = load_a32(arow, lane, 64);
  const v16h a3 = load_a32(arow, lane, 96);

  v8f acc[4];
#pragma unroll
  for (int t = 0; t < 4; t++) {
    const f16* brow = w16 + (size_t)((tg * 4 + t) * 16 + nl) * NC;
    v8f c = {};
    c = wmma32(a0, load_bt32(brow, lane, 0), c);
    c = wmma32(a1, load_bt32(brow, lane, 32), c);
    c = wmma32(a2, load_bt32(brow, lane, 64), c);
    c = wmma32(a3, load_bt32(brow, lane, 96), c);
    acc[t] = c;
  }

#pragma unroll
  for (int t = 0; t < 4; t++) {
    const int o = (tg * 4 + t) * 16 + nl;
    const float bv = bias[o];
#pragma unroll
    for (int r = 0; r < 8; r++) {
      int row = tm * 16 + r + 8 * h;
      int b   = row >> 12, hw = row & (NHW - 1);
      size_t idx = ((size_t)b * NC + o) * NHW + hw;
      out[idx] = acc[t][r] + bv + skip[idx];
    }
  }
}

// ---------------- launch ----------------

extern "C" void kernel_launch(void* const* d_in, const int* in_sizes, int n_in,
                              void* d_out, int out_size, void* d_ws, size_t ws_size,
                              hipStream_t stream) {
  const float* x      = (const float*)d_in[0];
  const float* norm_w = (const float*)d_in[1];
  const float* norm_b = (const float*)d_in[2];
  const float* qkv_w  = (const float*)d_in[3];
  const float* qkv_b  = (const float*)d_in[4];
  const float* proj_w = (const float*)d_in[5];
  const float* proj_b = (const float*)d_in[6];
  float* out = (float*)d_out;

  char* ws = (char*)d_ws;
  float* stats  = (float*)(ws);                    //   1 KB (256 f32)
  f16* qkvw16   = (f16*)(ws + 1024);               //  96 KB
  f16* projw16  = (f16*)(ws + 99328);              //  32 KB
  f16* xnt      = (f16*)(ws + 132096);             //   4 MB (16384x128)
  f16* qbuf     = (f16*)(ws + 4326400);            //   4 MB (16,4096,32)
  f16* kbuf     = (f16*)(ws + 8520704);            //   4 MB
  f16* vtbuf    = (f16*)(ws + 12715008);           //   4 MB (16,32,4096)
  f16* ot       = (f16*)(ws + 16909312);           //   4 MB (16384x128)

  cvt_f16_kernel<<<(3 * NC * NC + 255) / 256, 256, 0, stream>>>(qkv_w, qkvw16, 3 * NC * NC);
  cvt_f16_kernel<<<(NC * NC + 255) / 256, 256, 0, stream>>>(proj_w, projw16, NC * NC);
  gn_stats_kernel<<<NB * NGROUP, 256, 0, stream>>>(x, stats);
  gn_apply_kernel<<<(NB * NC * NHW) / 256, 256, 0, stream>>>(x, stats, norm_w, norm_b, xnt);
  // 1024 m-tiles * 6 n-groups, 4 waves / block
  qkv_gemm_kernel<<<(1024 * 6) / 4, 128, 0, stream>>>(xnt, qkvw16, qkv_b, qbuf, kbuf, vtbuf);
  // 16 (b,h) * 256 q tiles, 4 waves / block
  attn_kernel<<<(16 * 256) / 4, 128, 0, stream>>>(qbuf, kbuf, vtbuf, ot);
  // 1024 m-tiles * 2 n-groups, 4 waves / block
  proj_gemm_kernel<<<(1024 * 2) / 4, 128, 0, stream>>>(ot, projw16, proj_b, x, out);
}
